// SANet_22256520528170
// MI455X (gfx1250) — matrix operand order repaired and checked by reading, not compiled, not run on hardware
//
#include <hip/hip_runtime.h>
#include <hip/hip_bf16.h>
#include <stdint.h>

// ---------------------------------------------------------------------------
// SANet attention for MI455X (gfx1250): bf16 WMMA flash-attention pipeline.
// v2: vectorized (b128) fragment loads via shufflevector, pre-converted bf16
// weights, transposed LDS staging in the projection GEMMs.
// ---------------------------------------------------------------------------

typedef __bf16 bf16_t;
typedef __bf16 v16bf __attribute__((ext_vector_type(16)));
typedef __bf16 v8bf  __attribute__((ext_vector_type(8)));
typedef float  v8f   __attribute__((ext_vector_type(8)));

#define HW    4096
#define CDIM  512
#define BATCH 4
#define EPSI  1e-5f

// ---------------------------------------------------------------------------
// WMMA helpers (layouts per CDNA5 ISA §7.12.2, wave32)
// ---------------------------------------------------------------------------

__device__ __forceinline__ v8f wmma_bf16(v16bf a, v16bf b, v8f c) {
  return __builtin_amdgcn_wmma_f32_16x16x32_bf16(
      /*neg_a=*/false, a, /*neg_b=*/false, b,
      /*c_mod=*/(short)0, c, /*reuse_a=*/false, /*reuse_b=*/false);
}

__device__ __forceinline__ v16bf cat8(v8bf lo, v8bf hi) {
  return __builtin_shufflevector(lo, hi, 0, 1, 2, 3, 4, 5, 6, 7,
                                         8, 9, 10, 11, 12, 13, 14, 15);
}

// A-matrix fragment: 16(M) x 32(K) bf16, row-major source, row stride ld.
// lane<16 : row M=lane,    K in {0..7, 16..23}
// lane>=16: row M=lane-16, K in {8..15, 24..31}
// Per-lane data = two contiguous 8-element (16 B) runs -> two b128 loads.
// Requires (ld * 2) % 16 == 0 and 16B-aligned base (true for all call sites).
__device__ __forceinline__ v16bf load_a_frag(const bf16_t* src, int ld) {
  int lane = threadIdx.x & 31;
  int m  = lane & 15;
  int kb = (lane < 16) ? 0 : 8;
  const bf16_t* row = src + (size_t)m * ld + kb;
  v8bf lo = *(const v8bf*)(row);
  v8bf hi = *(const v8bf*)(row + 16);
  return cat8(lo, hi);
}

// B-matrix fragment: 32(K) x 16(N) where the SOURCE is stored [N][K] row-major
// with row stride ld (per-lane elements contiguous along K -> 2x b128 loads).
// lane<16 : col N=lane,    K in {0..7, 16..23} ; lane>=16: col N=lane-16, K in {8..15, 24..31}
__device__ __forceinline__ v16bf load_b_frag_t(const bf16_t* src, int ld) {
  int lane = threadIdx.x & 31;
  int n  = lane & 15;
  int kb = (lane < 16) ? 0 : 8;
  const bf16_t* p = src + (size_t)n * ld + kb;
  v8bf lo = *(const v8bf*)(p);
  v8bf hi = *(const v8bf*)(p + 16);
  return cat8(lo, hi);
}

// ---------------------------------------------------------------------------
// 0) One-shot fp32 -> bf16 weight conversion (512x512 per matrix).
// ---------------------------------------------------------------------------
__global__ void cvtw_kernel(const float* __restrict__ src,
                            bf16_t* __restrict__ dst) {
  int i = (blockIdx.x * blockDim.x + threadIdx.x) * 4;
#pragma unroll
  for (int j = 0; j < 4; ++j) dst[i + j] = (bf16_t)src[i + j];
}

// ---------------------------------------------------------------------------
// 1) Instance-norm statistics: mean / rstd per (b,c), var with ddof=1.
// ---------------------------------------------------------------------------
__global__ void stats_kernel(const float* __restrict__ content,
                             const float* __restrict__ style,
                             float* __restrict__ meanC, float* __restrict__ rstdC,
                             float* __restrict__ meanS, float* __restrict__ rstdS) {
  int c = blockIdx.x, b = blockIdx.y, which = blockIdx.z;
  const float* src = which ? style : content;
  const float* p = src + ((size_t)(b * CDIM + c)) * HW;
  float s1 = 0.f, s2 = 0.f;
  for (int i = threadIdx.x; i < HW; i += blockDim.x) {
    float v = p[i];
    s1 += v; s2 += v * v;
  }
#pragma unroll
  for (int m = 16; m >= 1; m >>= 1) {
    s1 += __shfl_xor(s1, m);
    s2 += __shfl_xor(s2, m);
  }
  __shared__ float ls1[8], ls2[8];
  int w = threadIdx.x >> 5;
  if ((threadIdx.x & 31) == 0) { ls1[w] = s1; ls2[w] = s2; }
  __syncthreads();
  if (threadIdx.x == 0) {
    float t1 = 0.f, t2 = 0.f;
    int nw = blockDim.x >> 5;
    for (int i = 0; i < nw; ++i) { t1 += ls1[i]; t2 += ls2[i]; }
    float mean = t1 / (float)HW;
    float var  = (t2 - (float)HW * mean * mean) / (float)(HW - 1) + EPSI;
    int idx = b * CDIM + c;
    if (which) { meanS[idx] = mean; rstdS[idx] = rsqrtf(var); }
    else       { meanC[idx] = mean; rstdC[idx] = rsqrtf(var); }
  }
}

// ---------------------------------------------------------------------------
// 2) Q/K/V projection GEMMs: out[o,x] = sum_c W[o,c] * normfeat[c,x] + bias[o]
//    mode 0: Q -> [b][x][c] bf16 (normalized content)
//    mode 1: K -> [b][x][c] bf16 (normalized style)
//    mode 2: V -> [b][c][x] bf16 (raw style, stored transposed)
//    Block tile = 64(o) x 32(x); 8 waves, one 16x16 WMMA tile each.
//    Feature tile staged TRANSPOSED in LDS ([x][c], row pad 40 -> 80 B rows)
//    so B-fragments are contiguous along K (2x ds_load_b128 per fragment).
// ---------------------------------------------------------------------------
__global__ void qkv_kernel(const float* __restrict__ feat,
                           const bf16_t* __restrict__ Wbf,   // bf16 weights
                           const float* __restrict__ bias,
                           const float* __restrict__ mean,
                           const float* __restrict__ rstd,
                           bf16_t* __restrict__ dst,
                           int mode) {
  __shared__ __align__(16) bf16_t Fl[32][40];   // [x][c] tile, 80B row stride
  int b  = blockIdx.z;
  int x0 = blockIdx.x * 32;
  int oB = blockIdx.y * 64;
  int w = threadIdx.x >> 5;
  int osub = w & 3, xsub = w >> 2;
  int obase = oB + osub * 16;
  const float* featB = feat + ((size_t)b * CDIM) * HW;

  v8f acc = {};
  for (int c0 = 0; c0 < CDIM; c0 += 32) {
    __syncthreads();
    {  // stage + normalize 32c x 32x feature tile into LDS, transposed
      int t  = threadIdx.x;
      int cc = t >> 3;             // 0..31 (channel within tile)
      int xx = (t & 7) * 4;        // 0..28 (spatial within tile)
      const float* gp = featB + (size_t)(c0 + cc) * HW + x0 + xx;
      float m_ = 0.f, r_ = 1.f;
      if (mode < 2) {
        int si = b * CDIM + c0 + cc;
        m_ = mean[si]; r_ = rstd[si];
      }
#pragma unroll
      for (int j = 0; j < 4; ++j) Fl[xx + j][cc] = (bf16_t)((gp[j] - m_) * r_);
    }
    __syncthreads();
    v16bf a  = load_a_frag(Wbf + (size_t)obase * CDIM + c0, CDIM);
    v16bf bb = load_b_frag_t(&Fl[xsub * 16][0], 40);
    acc = wmma_bf16(a, bb, acc);
  }

  int ln = threadIdx.x & 31;
  int n = ln & 15, half = (ln >> 4) & 1;
  int x = x0 + xsub * 16 + n;
#pragma unroll
  for (int r = 0; r < 8; ++r) {
    int o = obase + r + 8 * half;
    float v = acc[r] + bias[o];
    if (mode < 2) dst[((size_t)(b * HW + x)) * CDIM + o] = (bf16_t)v;
    else          dst[((size_t)(b * CDIM + o)) * HW + x] = (bf16_t)v;
  }
}

// ---------------------------------------------------------------------------
// 3) Single-pass flash attention.
//    Block: 256 threads = 8 waves, q-tile = 32 rows (2 q-subtiles x 16).
//    wave w: qs = w&1 (its q-subtile), ch = w>>1 (its 128-channel V slice AND
//    its pair of 16-wide score sub-tiles per 128-s superstep).
//    Per superstep each wave: 32 score WMMAs + 32 PV WMMAs (balanced).
// ---------------------------------------------------------------------------
__global__ void attn_kernel(const bf16_t* __restrict__ Qg,
                            const bf16_t* __restrict__ Kg,
                            const bf16_t* __restrict__ Vt,
                            bf16_t* __restrict__ Ows) {
  __shared__ __align__(16) bf16_t Qs[32][CDIM];   // 32 KB, cached Q tile
  __shared__ __align__(16) bf16_t Pl[2][16][128]; // 8 KB, P for this superstep
  __shared__ float redmax[2][4][16];
  __shared__ float redsum[2][4][16];

  int b  = blockIdx.y;
  int q0 = blockIdx.x * 32;
  int w  = threadIdx.x >> 5;
  int ln = threadIdx.x & 31;
  int qs = w & 1;
  int ch = w >> 1;                 // 0..3
  int half = (ln >> 4) & 1;
  int n = ln & 15;

  const bf16_t* Qb = Qg + ((size_t)(b * HW + q0)) * CDIM;
  const bf16_t* Kb = Kg + ((size_t)b * HW) * CDIM;
  const bf16_t* Vb = Vt + ((size_t)b * CDIM) * HW;

  { // stage Q tile (32 x 512 bf16, fully contiguous) via 16B vectors
    const uint4* src  = (const uint4*)Qb;
    uint4*       dstl = (uint4*)&Qs[0][0];
    const int total = 32 * (CDIM / 8);
    for (int i = threadIdx.x; i < total; i += blockDim.x) dstl[i] = src[i];
  }
  __syncthreads();

  float m_old[8], l_run[8];
#pragma unroll
  for (int r = 0; r < 8; ++r) { m_old[r] = -3.0e38f; l_run[r] = 0.f; }
  v8f O[8];
#pragma unroll
  for (int t = 0; t < 8; ++t) O[t] = (v8f){};

  for (int sb = 0; sb < HW; sb += 128) {
    // ---- scores: two 16x16 tiles for s-subtiles (2ch, 2ch+1) -------------
    v8f S0 = {}, S1 = {};
    int s0 = sb + (2 * ch) * 16;
    int s1 = s0 + 16;
#pragma unroll 4
    for (int kk = 0; kk < CDIM; kk += 32) {
      v16bf a  = load_a_frag(&Qs[qs * 16][kk], CDIM);
      v16bf b0 = load_b_frag_t(Kb + (size_t)s0 * CDIM + kk, CDIM);
      v16bf b1 = load_b_frag_t(Kb + (size_t)s1 * CDIM + kk, CDIM);
      S0 = wmma_bf16(a, b0, S0);
      S1 = wmma_bf16(a, b1, S1);
    }
    // ---- per-row max over this wave's 32 s-columns (butterfly in half) ---
    float tm[8];
#pragma unroll
    for (int r = 0; r < 8; ++r) tm[r] = fmaxf(S0[r], S1[r]);
#pragma unroll
    for (int r = 0; r < 8; ++r)
      for (int msk = 8; msk >= 1; msk >>= 1)
        tm[r] = fmaxf(tm[r], __shfl_xor(tm[r], msk));

    __syncthreads();                         // prior-step P fully consumed
    if (n == 0) {
#pragma unroll
      for (int r = 0; r < 8; ++r) redmax[qs][ch][r + 8 * half] = tm[r];
    }
    __syncthreads();                         // redmax visible

    // ---- online softmax state update -------------------------------------
    float m_new[8], scale[8];
#pragma unroll
    for (int r = 0; r < 8; ++r) {
      int row = r + 8 * half;
      float mv = m_old[r];
      for (int cc = 0; cc < 4; ++cc) mv = fmaxf(mv, redmax[qs][cc][row]);
      m_new[r] = mv;
      scale[r] = __expf(m_old[r] - mv);
      l_run[r] *= scale[r];
      m_old[r] = mv;
    }
#pragma unroll
    for (int t = 0; t < 8; ++t)
#pragma unroll
      for (int r = 0; r < 8; ++r) O[t][r] *= scale[r];

    // ---- P = exp(S - m), partial row sums, publish P (bf16) to LDS -------
    float ps[8];
#pragma unroll
    for (int r = 0; r < 8; ++r) {
      float p0 = __expf(S0[r] - m_new[r]);
      float p1 = __expf(S1[r] - m_new[r]);
      Pl[qs][r + 8 * half][32 * ch + n]      = (bf16_t)p0;
      Pl[qs][r + 8 * half][32 * ch + 16 + n] = (bf16_t)p1;
      ps[r] = p0 + p1;
    }
#pragma unroll
    for (int r = 0; r < 8; ++r)
      for (int msk = 8; msk >= 1; msk >>= 1) ps[r] += __shfl_xor(ps[r], msk);
    if (n == 0) {
#pragma unroll
      for (int r = 0; r < 8; ++r) redsum[qs][ch][r + 8 * half] = ps[r];
    }
    __syncthreads();                         // P + redsum visible

#pragma unroll
    for (int r = 0; r < 8; ++r) {
      int row = r + 8 * half;
      float sadd = 0.f;
      for (int cc = 0; cc < 4; ++cc) sadd += redsum[qs][cc][row];
      l_run[r] += sadd;
    }

    // ---- O += P(16x128) x V(128 x 128-channel slice) ---------------------
#pragma unroll
    for (int kk2 = 0; kk2 < 4; ++kk2) {
      v16bf a = load_a_frag(&Pl[qs][0][32 * kk2], 128);
#pragma unroll
      for (int t = 0; t < 8; ++t) {
        int c0 = ch * 128 + t * 16;
        v16bf bb = load_b_frag_t(Vb + (size_t)c0 * HW + sb + 32 * kk2, HW);
        O[t] = wmma_bf16(a, bb, O[t]);
      }
    }

    if (sb + 128 < HW) {   // hint next K/V superstep into cache
      __builtin_prefetch(Kb + (size_t)(sb + 128 + ch * 32) * CDIM, 0, 1);
      __builtin_prefetch(Vb + (size_t)(ch * 128) * HW + sb + 128, 0, 1);
    }
  }

  // ---- finalize: O /= l, store bf16 [b][q][c] ----------------------------
#pragma unroll
  for (int r = 0; r < 8; ++r) {
    float inv = 1.0f / l_run[r];
#pragma unroll
    for (int t = 0; t < 8; ++t) O[t][r] *= inv;
  }
#pragma unroll
  for (int t = 0; t < 8; ++t) {
    int c = ch * 128 + t * 16 + n;
#pragma unroll
    for (int r = 0; r < 8; ++r) {
      int q = q0 + qs * 16 + r + 8 * half;
      Ows[((size_t)(b * HW + q)) * CDIM + c] = (bf16_t)O[t][r];
    }
  }
}

// ---------------------------------------------------------------------------
// 4) out[b][o][q] = sum_c Wo[o][c] * O[b][q][c] + bo[o] + content[b][o][q]
// ---------------------------------------------------------------------------
__global__ void out_kernel(const bf16_t* __restrict__ Ows,
                           const bf16_t* __restrict__ Wobf,   // bf16 weights
                           const float* __restrict__ bo,
                           const float* __restrict__ content,
                           float* __restrict__ out) {
  int b  = blockIdx.z;
  int q0 = blockIdx.x * 32;
  int oB = blockIdx.y * 64;
  int w = threadIdx.x >> 5;
  int osub = w & 3, qsub = w >> 2;
  int obase = oB + osub * 16;
  int qbase = q0 + qsub * 16;
  int ln = threadIdx.x & 31;
  int n = ln & 15, half = (ln >> 4) & 1;

  const bf16_t* Ob = Ows + ((size_t)(b * HW + qbase)) * CDIM;
  v8f acc = {};
#pragma unroll 4
  for (int c0 = 0; c0 < CDIM; c0 += 32) {
    v16bf a  = load_a_frag(Wobf + (size_t)obase * CDIM + c0, CDIM);
    v16bf bb = load_b_frag_t(Ob + c0, CDIM);
    acc = wmma_bf16(a, bb, acc);
  }
  int q = qbase + n;
#pragma unroll
  for (int r = 0; r < 8; ++r) {
    int o = obase + r + 8 * half;
    size_t idx = ((size_t)(b * CDIM + o)) * HW + q;
    out[idx] = acc[r] + bo[o] + content[idx];
  }
}

// ---------------------------------------------------------------------------
// Launcher
// ---------------------------------------------------------------------------
extern "C" void kernel_launch(void* const* d_in, const int* in_sizes, int n_in,
                              void* d_out, int out_size, void* d_ws, size_t ws_size,
                              hipStream_t stream) {
  (void)in_sizes; (void)n_in; (void)out_size; (void)ws_size;
  const float* content = (const float*)d_in[0];
  const float* style   = (const float*)d_in[1];
  const float* Wf  = (const float*)d_in[2];
  const float* bf_ = (const float*)d_in[3];
  const float* Wg  = (const float*)d_in[4];
  const float* bg  = (const float*)d_in[5];
  const float* Wh  = (const float*)d_in[6];
  const float* bh  = (const float*)d_in[7];
  const float* Wo  = (const float*)d_in[8];
  const float* bo  = (const float*)d_in[9];
  float* out = (float*)d_out;

  char* ws = (char*)d_ws;
  size_t off = 0;
  auto carve = [&](size_t bytes) -> void* {
    void* p = ws + off;
    off += (bytes + 255) & ~(size_t)255;
    return p;
  };
  float* meanC = (float*)carve((size_t)BATCH * CDIM * sizeof(float));
  float* rstdC = (float*)carve((size_t)BATCH * CDIM * sizeof(float));
  float* meanS = (float*)carve((size_t)BATCH * CDIM * sizeof(float));
  float* rstdS = (float*)carve((size_t)BATCH * CDIM * sizeof(float));
  size_t wmat_bytes = (size_t)CDIM * CDIM * sizeof(bf16_t);        // 512 KB
  bf16_t* Wf16 = (bf16_t*)carve(wmat_bytes);
  bf16_t* Wg16 = (bf16_t*)carve(wmat_bytes);
  bf16_t* Wh16 = (bf16_t*)carve(wmat_bytes);
  bf16_t* Wo16 = (bf16_t*)carve(wmat_bytes);
  size_t qkv_bytes = (size_t)BATCH * HW * CDIM * sizeof(bf16_t);   // 16 MB
  bf16_t* Qb   = (bf16_t*)carve(qkv_bytes);
  bf16_t* Kb   = (bf16_t*)carve(qkv_bytes);
  bf16_t* Vtb  = (bf16_t*)carve(qkv_bytes);
  bf16_t* Owsb = (bf16_t*)carve(qkv_bytes);

  // 0) weights -> bf16 (once per launch; tiny)
  dim3 gCvt(CDIM * CDIM / (256 * 4));
  cvtw_kernel<<<gCvt, 256, 0, stream>>>(Wf, Wf16);
  cvtw_kernel<<<gCvt, 256, 0, stream>>>(Wg, Wg16);
  cvtw_kernel<<<gCvt, 256, 0, stream>>>(Wh, Wh16);
  cvtw_kernel<<<gCvt, 256, 0, stream>>>(Wo, Wo16);

  // 1) instance-norm stats for content + style
  stats_kernel<<<dim3(CDIM, BATCH, 2), 256, 0, stream>>>(
      content, style, meanC, rstdC, meanS, rstdS);

  // 2) projections (bf16 Q/K and transposed V)
  dim3 gGemm(HW / 32, CDIM / 64, BATCH);
  qkv_kernel<<<gGemm, 256, 0, stream>>>(content, Wf16, bf_, meanC, rstdC, Qb, 0);
  qkv_kernel<<<gGemm, 256, 0, stream>>>(style,   Wg16, bg,  meanS, rstdS, Kb, 1);
  qkv_kernel<<<gGemm, 256, 0, stream>>>(style,   Wh16, bh,  meanS, rstdS, Vtb, 2);

  // 3) flash attention (single pass, online softmax)
  attn_kernel<<<dim3(HW / 32, BATCH), 256, 0, stream>>>(Qb, Kb, Vtb, Owsb);

  // 4) output projection + residual
  out_kernel<<<gGemm, 256, 0, stream>>>(Owsb, Wo16, bo, content, out);
}
